// MPEDecoder_83434034692196
// MI455X (gfx1250) — matrix operand order repaired and checked by reading, not compiled, not run on hardware
//
#include <hip/hip_runtime.h>
#include <hip/hip_bf16.h>
#include <math.h>

// Problem constants (fixed by the reference)
#define B_ 16
#define N_ 500
#define D_ 32
#define H_ 128
#define P_ 100
#define E_ (N_ * D_)   // 16000

typedef __attribute__((ext_vector_type(16))) _Float16 v16h;
typedef __attribute__((ext_vector_type(8)))  float    v8f;

// ---------------------------------------------------------------------------
// WMMA helpers (CDNA5 16x16x32 f16 -> f32). Fragment layouts per ISA 7.12.2:
//  A 16x32: lanes 0-15 M=0..15; halves 0..7 K=hi*8+0..7, halves 8..15 K=16+hi*8+..
//  B 32x16: lane n=L&15; halves 0..7 K=hi*8+0..7, halves 8..15 K=16+hi*8+..
//  C/D 16x16: lane n=L&15; vgpr v: M = hi*8 + v
// ---------------------------------------------------------------------------
__device__ __forceinline__ v8f wmma_f16(v16h a, v16h b, v8f c) {
  return __builtin_amdgcn_wmma_f32_16x16x32_f16(false, a, false, b, (short)0, c,
                                                false, false);
}

__device__ __forceinline__ v16h load_a_f16(const _Float16* A, int lane, int mbase,
                                           int kbase, int lda) {
  int m  = mbase + (lane & 15);
  int hi = lane >> 4;
  v16h a;
#pragma unroll
  for (int t = 0; t < 8; ++t) a[t] = A[(size_t)m * lda + kbase + hi * 8 + t];
#pragma unroll
  for (int t = 0; t < 8; ++t) a[8 + t] = A[(size_t)m * lda + kbase + 16 + hi * 8 + t];
  return a;
}

__device__ __forceinline__ v16h load_b_f16(const _Float16* B, int lane, int kbase,
                                           int nbase, int ldb) {
  int n  = nbase + (lane & 15);
  int hi = lane >> 4;
  v16h b;
#pragma unroll
  for (int t = 0; t < 16; ++t) b[t] = B[(size_t)(kbase + (t >> 3) * 16 + hi * 8 + (t & 7)) * ldb + n];
  return b;
}

// A[i,k] = W[k,i] (transposed read from f32), for Mw = Wk^T @ Wq
__device__ __forceinline__ v16h load_a_trans_f32(const float* W, int lane, int mbase,
                                                 int kbase, int ld) {
  int i  = mbase + (lane & 15);
  int hi = lane >> 4;
  v16h a;
#pragma unroll
  for (int t = 0; t < 8; ++t) a[t] = (_Float16)W[(size_t)(kbase + hi * 8 + t) * ld + i];
#pragma unroll
  for (int t = 0; t < 8; ++t)
    a[8 + t] = (_Float16)W[(size_t)(kbase + 16 + hi * 8 + t) * ld + i];
  return a;
}

__device__ __forceinline__ v16h load_b_f32(const float* B, int lane, int kbase,
                                           int nbase, int ldb) {
  int n  = nbase + (lane & 15);
  int hi = lane >> 4;
  v16h b;
#pragma unroll
  for (int t = 0; t < 16; ++t) b[t] = (_Float16)B[(size_t)(kbase + (t >> 3) * 16 + hi * 8 + (t & 7)) * ldb + n];
  return b;
}

// ---------------------------------------------------------------------------
// Kernel 0: zero the embedding-sum accumulator (ws is not re-zeroed by harness)
// ---------------------------------------------------------------------------
__global__ void zero_kernel(float* p, int n) {
  int i = blockIdx.x * blockDim.x + threadIdx.x;
  if (i < n) p[i] = 0.f;
}

// ---------------------------------------------------------------------------
// Kernel 1: S[b,h] = sum_e emb[b,e,h].  16x125 blocks, coalesced 512B rows.
// ---------------------------------------------------------------------------
__global__ __launch_bounds__(128) void embsum_kernel(const float* __restrict__ emb,
                                                     float* __restrict__ S) {
  int b  = blockIdx.x;
  int e0 = blockIdx.y * 128;
  int t  = threadIdx.x;
  float acc = 0.f;
#pragma unroll 4
  for (int i = 0; i < 128; ++i)
    acc += emb[((size_t)b * E_ + e0 + i) * H_ + t];
  atomicAdd(&S[b * H_ + t], acc);
}

// ---------------------------------------------------------------------------
// Kernel 2: weight prep.
//  WcatT[k,n] (f16, 256x128): k<128 -> (Wq_last + Wq_visited/N)^T ; k>=128 -> Wq_visited^T
//  MwT[k,n]  (f16, 128x128): MwT[j,i] = Mw[i,j], Mw = Wk^T @ Wq   (WMMA)
// ---------------------------------------------------------------------------
__global__ __launch_bounds__(256) void wprep_kernel(const float* __restrict__ Wq_last,
                                                    const float* __restrict__ Wq_visited,
                                                    const float* __restrict__ Wq,
                                                    const float* __restrict__ Wk,
                                                    _Float16* __restrict__ WcatT,
                                                    _Float16* __restrict__ MwT) {
  int tid = threadIdx.x;
  for (int idx = tid; idx < 256 * 128; idx += 256) {
    int k = idx >> 7, n = idx & 127;
    float v;
    if (k < 128)
      v = Wq_last[n * 128 + k] + Wq_visited[n * 128 + k] * (1.0f / (float)N_);
    else
      v = Wq_visited[n * 128 + (k - 128)];
    WcatT[idx] = (_Float16)v;
  }
  // Mw = Wk^T @ Wq via WMMA: 8x8 tiles of 16x16, K=128 in 4 steps
  int w = tid >> 5, lane = tid & 31;
  int l15 = lane & 15, hi = lane >> 4;
  v8f zero = {0.f, 0.f, 0.f, 0.f, 0.f, 0.f, 0.f, 0.f};
  for (int t8 = 0; t8 < 8; ++t8) {
    int tile = w * 8 + t8;
    int mt = tile >> 3, nt = tile & 7;
    v8f c = zero;
#pragma unroll
    for (int s = 0; s < 4; ++s) {
      v16h a = load_a_trans_f32(Wk, lane, mt * 16, s * 32, 128);
      v16h b = load_b_f32(Wq, lane, s * 32, nt * 16, 128);
      c = wmma_f16(a, b, c);
    }
#pragma unroll
    for (int v = 0; v < 8; ++v) {
      int i = mt * 16 + hi * 8 + v;  // Mw row
      int j = nt * 16 + l15;         // Mw col
      MwT[j * 128 + i] = (_Float16)c[v];
    }
  }
}

// ---------------------------------------------------------------------------
// Kernel 3: per-batch projection chain (all WMMA):
//  fq = q_first + q_graph + le @ Wc^T + ve @ Wqv^T   (K=256 concat GEMM)
//  qtilde = fq @ Mw^T
// One block per b, 8 waves; 7 M-tiles (P=100 padded to 112) x 8 N-tiles.
// ---------------------------------------------------------------------------
__global__ __launch_bounds__(256) void proj_kernel(const float* __restrict__ S,
                                                   const float* __restrict__ Wq_graph,
                                                   const float* __restrict__ q_first,
                                                   const float* __restrict__ ve,
                                                   const float* __restrict__ le,
                                                   const _Float16* __restrict__ WcatT,
                                                   const _Float16* __restrict__ MwT,
                                                   float* __restrict__ qtilde) {
  __shared__ _Float16 sA[112 * 128];
  __shared__ _Float16 sFq[112 * 128];
  __shared__ float    sQg[128];

  int b   = blockIdx.x;
  int tid = threadIdx.x;

  // q_graph[h] = (sum_e emb / E) . Wq_graph[h,:]
  if (tid < 128) {
    float acc = 0.f;
    for (int j = 0; j < 128; ++j)
      acc += (S[b * 128 + j] * (1.0f / (float)E_)) * Wq_graph[tid * 128 + j];
    sQg[tid] = acc;
  }

  int w = tid >> 5, lane = tid & 31, l15 = lane & 15, hi = lane >> 4;
  v8f zero = {0.f, 0.f, 0.f, 0.f, 0.f, 0.f, 0.f, 0.f};
  v8f c[7];
#pragma unroll
  for (int mt = 0; mt < 7; ++mt) c[mt] = zero;

  // two K-halves: pass0 A=last_edge (Wc has /N folded), pass1 A=visited
  for (int pass = 0; pass < 2; ++pass) {
    const float* src = pass ? ve : le;
    __syncthreads();
    for (int idx = tid; idx < 112 * 128; idx += 256) {
      int m = idx >> 7, j = idx & 127;
      sA[idx] = (m < P_) ? (_Float16)src[((size_t)b * P_ + m) * H_ + j] : (_Float16)0.f;
    }
    __syncthreads();
    for (int mt = 0; mt < 7; ++mt) {
#pragma unroll
      for (int s = 0; s < 4; ++s) {
        v16h a  = load_a_f16(sA, lane, mt * 16, s * 32, 128);
        v16h bb = load_b_f16(WcatT, lane, pass * 128 + s * 32, w * 16, 128);
        c[mt] = wmma_f16(a, bb, c[mt]);
      }
    }
  }
  __syncthreads();

  // fq = acc + q_graph + q_first  -> f16 in LDS for the second GEMM
  for (int mt = 0; mt < 7; ++mt) {
#pragma unroll
    for (int v = 0; v < 8; ++v) {
      int m = mt * 16 + hi * 8 + v;
      int n = w * 16 + l15;
      float val = c[mt][v] + sQg[n];
      if (m < P_) val += q_first[((size_t)b * P_ + m) * H_ + n];
      sFq[m * 128 + n] = (_Float16)val;
    }
  }
  __syncthreads();

  // qtilde = fq @ Mw^T
  for (int mt = 0; mt < 7; ++mt) {
    v8f c2 = zero;
#pragma unroll
    for (int s = 0; s < 4; ++s) {
      v16h a  = load_a_f16(sFq, lane, mt * 16, s * 32, 128);
      v16h bb = load_b_f16(MwT, lane, s * 32, w * 16, 128);
      c2 = wmma_f16(a, bb, c2);
    }
#pragma unroll
    for (int v = 0; v < 8; ++v) {
      int m = mt * 16 + hi * 8 + v;
      int n = w * 16 + l15;
      if (m < P_) qtilde[((size_t)b * P_ + m) * H_ + n] = c2[v];
    }
  }
}

// ---------------------------------------------------------------------------
// Kernel 4: per-(b,p): gather 32 outgoing rows (float4, also written to output),
// 128-dot vs qtilde via wave32 shuffle reduce, dist/mask fixups, tanh clip,
// 32-wide softmax in one wave.
// ---------------------------------------------------------------------------
__global__ __launch_bounds__(128) void score_kernel(const float* __restrict__ emb,
                                                    const float* __restrict__ dists,
                                                    const float* __restrict__ pref,
                                                    const int* __restrict__ indices,
                                                    const int* __restrict__ lni,
                                                    const float* __restrict__ gmask,
                                                    const float* __restrict__ qtilde,
                                                    float* __restrict__ probs,
                                                    int* __restrict__ nk_out,
                                                    float* __restrict__ oe_out) {
  __shared__ float sQt[128];
  __shared__ float sSc[32];
  __shared__ int   sNk[32];

  int bp = blockIdx.x;
  int b = bp / P_, p = bp % P_;
  int tid = threadIdx.x;
  if (tid < 128) sQt[tid] = qtilde[(size_t)bp * H_ + tid];
  int ln = lni[bp];
  __syncthreads();

  int w = tid >> 5, lane = tid & 31;
  const float p0 = pref[0], p1 = pref[1];
  const float SCALE = 1.0f / (8.0f * 11.313708498984761f);  // 1/(8*sqrt(128))
  const float RS2   = 0.70710678118654752440f;              // 1/sqrt(2)

  for (int dd = 0; dd < 8; ++dd) {
    int d = w * 8 + dd;
    size_t oi = (size_t)ln * D_ + d;
    const float4 ev = *(const float4*)(emb + ((size_t)b * E_ + oi) * H_ + lane * 4);
    *(float4*)(oe_out + ((size_t)bp * D_ + d) * H_ + lane * 4) = ev;
    float part = ev.x * sQt[lane * 4 + 0] + ev.y * sQt[lane * 4 + 1] +
                 ev.z * sQt[lane * 4 + 2] + ev.w * sQt[lane * 4 + 3];
#pragma unroll
    for (int off = 16; off > 0; off >>= 1) part += __shfl_xor(part, off, 32);
    if (lane == 0) {
      int nk = indices[(((size_t)b * E_) + oi) * 2 + 1];
      const float* dp = dists + (((size_t)b * N_ + ln) * N_ + nk) * 2;
      float dc = p0 * dp[0] + p1 * dp[1];
      float mk = gmask[((size_t)b * P_ + p) * N_ + nk];
      if (mk < -3.0e38f) mk = -1.0e8f;  // isneginf -> -1e8
      sSc[d] = 10.f * tanhf(part * SCALE - dc * RS2) + mk;
      sNk[d] = nk;
    }
  }
  __syncthreads();

  // softmax over D=32 in one wave32
  if (tid < 32) {
    float sc = sSc[tid];
    float mx = sc;
#pragma unroll
    for (int off = 16; off > 0; off >>= 1) mx = fmaxf(mx, __shfl_xor(mx, off, 32));
    float e = __expf(sc - mx);
    float sum = e;
#pragma unroll
    for (int off = 16; off > 0; off >>= 1) sum += __shfl_xor(sum, off, 32);
    probs[(size_t)bp * D_ + tid] = e / sum;
    nk_out[(size_t)bp * D_ + tid] = sNk[tid];
  }
}

// ---------------------------------------------------------------------------
// Host launcher
// ---------------------------------------------------------------------------
extern "C" void kernel_launch(void* const* d_in, const int* in_sizes, int n_in,
                              void* d_out, int out_size, void* d_ws, size_t ws_size,
                              hipStream_t stream) {
  (void)in_sizes; (void)n_in; (void)out_size; (void)ws_size;

  const float* emb        = (const float*)d_in[0];
  const float* dists      = (const float*)d_in[1];
  const float* pref       = (const float*)d_in[2];
  const int*   indices    = (const int*)d_in[3];
  /* d_in[4] = Wq_first — unused by the reference */
  const float* Wq_last    = (const float*)d_in[5];
  const float* Wq_visited = (const float*)d_in[6];
  const float* Wq_graph   = (const float*)d_in[7];
  const float* Wq         = (const float*)d_in[8];
  const float* Wk         = (const float*)d_in[9];
  const float* q_first    = (const float*)d_in[10];
  const float* ve         = (const float*)d_in[11];
  const float* le         = (const float*)d_in[12];
  const int*   lni        = (const int*)d_in[13];
  const float* gmask      = (const float*)d_in[14];

  // workspace layout (bytes)
  char* ws = (char*)d_ws;
  float*    S      = (float*)(ws);                 //  8,192 B  (B*H f32)
  float*    qtilde = (float*)(ws + 8192);          // 819,200 B (B*P*H f32)
  _Float16* WcatT  = (_Float16*)(ws + 827392);     //  65,536 B (256*128 f16)
  _Float16* MwT    = (_Float16*)(ws + 892928);     //  32,768 B (128*128 f16)

  // output layout: probs (B*P*D f32) | node_keys (B*P*D i32) | outgoing (B*P*D*H f32)
  float* probs = (float*)d_out;
  int*   nko   = (int*)(probs + B_ * P_ * D_);
  float* oe    = probs + 2 * (B_ * P_ * D_);

  zero_kernel<<<dim3(8), 256, 0, stream>>>(S, B_ * H_);
  embsum_kernel<<<dim3(B_, 125), 128, 0, stream>>>(emb, S);
  wprep_kernel<<<1, 256, 0, stream>>>(Wq_last, Wq_visited, Wq, Wk, WcatT, MwT);
  proj_kernel<<<B_, 256, 0, stream>>>(S, Wq_graph, q_first, ve, le, WcatT, MwT, qtilde);
  score_kernel<<<B_ * P_, 128, 0, stream>>>(emb, dists, pref, indices, lni, gmask,
                                            qtilde, probs, nko, oe);
}
